// GCL_17308718202949
// MI455X (gfx1250) — matrix operand order repaired and checked by reading, not compiled, not run on hardware
//
#include <hip/hip_runtime.h>
#include <cstdint>
#include <cstddef>

// Problem constants (from the reference)
#define IN_DIM  512
#define HID_DIM 256
#define OUT_DIM 128

// ---------- CDNA5 WMMA vector types (wave32) ----------
typedef __attribute__((ext_vector_type(16))) __bf16 v16bf;
typedef __attribute__((ext_vector_type(8)))  float  v8f;

static __device__ __forceinline__ uint16_t f2bf(float f) {
  union { float f; uint32_t u; } v; v.f = f;
  uint32_t u = v.u;
  uint32_t r = u + 0x7FFFu + ((u >> 16) & 1u);   // round-to-nearest-even
  return (uint16_t)(r >> 16);
}

// =====================================================================
// GEMM: C[M,Nc] = A[M,K](bf16, row-major) x Wt[Nc,K](bf16, pre-transposed)
// via v_wmma_f32_16x16x32_bf16.
//   Block = 256 thr = 8 wave32, block tile 64(M) x 128(N) x 32(K),
//   each wave owns a 32x32 output (2x2 WMMA accumulators).
//   Tiles are streamed global->LDS with GLOBAL_LOAD_ASYNC_TO_LDS_B128
//   (ASYNCcnt), double-buffered: next tile in flight while WMMAs consume
//   the current one; pipelined s_wait_asynccnt 0x3 (3 async ops/wave/tile).
//   Because Wt is pre-transposed, both A and B fragments are contiguous
//   16B ds_load_b128 reads matching the ISA 16-bit A(16x32)/B(32x16)
//   VGPR layouts (lane halves split K).
// =====================================================================
#define BM 64
#define BN 128
#define KC 32
#define LDT 40   // LDS row stride in bf16 elems: 80B, 16B-aligned, bank pad

__global__ __launch_bounds__(256)
void GCL_gemm_wmma_bf16(const uint16_t* __restrict__ A,   // bf16 [M,K]
                        const uint16_t* __restrict__ Wt,  // bf16 [Nc,K]
                        float* __restrict__ C,
                        const float* __restrict__ bias,
                        int M, int K, int Nc)
{
  __shared__ __align__(16) uint16_t As[2][BM * LDT];
  __shared__ __align__(16) uint16_t Bs[2][BN * LDT];   // [n][k], k contiguous

  const int tid  = threadIdx.x;
  const int lane = tid & 31;
  const int wave = tid >> 5;       // 0..7
  const int wm   = wave >> 2;      // 0..1  (32-row strip)
  const int wn   = wave & 3;       // 0..3  (32-col strip)
  const int half = lane >> 4;      // 0..1  (K/M halves per ISA layout)
  const int l16  = lane & 15;

  const int row0 = blockIdx.x * BM;
  const int col0 = blockIdx.y * BN;

  v8f acc[2][2];
#pragma unroll
  for (int i = 0; i < 2; ++i)
#pragma unroll
    for (int j = 0; j < 2; ++j)
#pragma unroll
      for (int k = 0; k < 8; ++k) acc[i][j][k] = 0.0f;

  // per-lane async-copy assignments (8 bf16 = 16B per async lane-slot)
  const int ar = tid >> 2;          // 0..63  : A row in tile
  const int ac = (tid & 3) * 8;     // 0..24  : A k-chunk
  const int bn = tid >> 2;          // 0..63  : B row (n) in tile, set 0
  const int bc = (tid & 3) * 8;     // 0..24  : B k-chunk

  const uint32_t ldsA = (uint32_t)(uintptr_t)(&As[0][0]);
  const uint32_t ldsB = (uint32_t)(uintptr_t)(&Bs[0][0]);

  // one async b128 per lane slot: per wave => 1 (A) + 2 (B) = 3 async ops
  auto issue = [&](int buf, int k0) {
    uint32_t la = ldsA + (uint32_t)((buf * BM + ar) * LDT + ac) * 2u;
    const uint16_t* ga = A + (size_t)(row0 + ar) * K + (k0 + ac);
    asm volatile("global_load_async_to_lds_b128 %0, %1, off"
                 :: "v"(la), "v"(ga) : "memory");
    uint32_t lb0 = ldsB + (uint32_t)((buf * BN + bn) * LDT + bc) * 2u;
    const uint16_t* gb0 = Wt + (size_t)(col0 + bn) * K + (k0 + bc);
    asm volatile("global_load_async_to_lds_b128 %0, %1, off"
                 :: "v"(lb0), "v"(gb0) : "memory");
    uint32_t lb1 = lb0 + (uint32_t)(64 * LDT) * 2u;
    const uint16_t* gb1 = gb0 + (size_t)64 * K;
    asm volatile("global_load_async_to_lds_b128 %0, %1, off"
                 :: "v"(lb1), "v"(gb1) : "memory");
  };

  const int nt = K / KC;
  issue(0, 0);

  for (int it = 0; it < nt; ++it) {
    const int buf = it & 1;
    if (it + 1 < nt) {
      issue(buf ^ 1, (it + 1) * KC);                 // prefetch next tile
      asm volatile("s_wait_asynccnt 0x3" ::: "memory");  // current tile done
    } else {
      asm volatile("s_wait_asynccnt 0x0" ::: "memory");
    }
    __syncthreads();   // all waves' async writes for tile `it` visible

    // ---- fragments per ISA 7.12.2 layouts ----
    // A 16x32 bf16: e=0..7 -> k=8h+0..7 ; e=8..15 -> k=16+8h+0..7
    // B 32x16 bf16: same k order as A: e=0..7 -> k=8h+0..7 ; e=8..15 -> k=16+8h+0..7
    union Frag { uint4 q[2]; v16bf v; };
    Frag a[2], b[2];
#pragma unroll
    for (int mt = 0; mt < 2; ++mt) {
      const uint16_t* ap = &As[buf][(wm * 32 + mt * 16 + l16) * LDT];
      a[mt].q[0] = *reinterpret_cast<const uint4*>(ap + 8 * half);
      a[mt].q[1] = *reinterpret_cast<const uint4*>(ap + 16 + 8 * half);
    }
#pragma unroll
    for (int nt2 = 0; nt2 < 2; ++nt2) {
      const uint16_t* bp = &Bs[buf][(wn * 32 + nt2 * 16 + l16) * LDT];
      b[nt2].q[0] = *reinterpret_cast<const uint4*>(bp + 8 * half);
      b[nt2].q[1] = *reinterpret_cast<const uint4*>(bp + 16 + 8 * half);
    }
#pragma unroll
    for (int mt = 0; mt < 2; ++mt)
#pragma unroll
      for (int nt2 = 0; nt2 < 2; ++nt2)
        acc[mt][nt2] = __builtin_amdgcn_wmma_f32_16x16x32_bf16(
            false, a[mt].v, false, b[nt2].v, (short)0, acc[mt][nt2],
            false, false);

    __syncthreads();   // done reading `buf` before it is refilled at it+2
  }

  // ---- epilogue: C layout = VGPR j -> row j+8*half, col = l16 ----
#pragma unroll
  for (int mt = 0; mt < 2; ++mt) {
#pragma unroll
    for (int nt2 = 0; nt2 < 2; ++nt2) {
      const int col = col0 + wn * 32 + nt2 * 16 + l16;
      const float bv = bias ? bias[col] : 0.0f;
#pragma unroll
      for (int j = 0; j < 8; ++j) {
        const int row = row0 + wm * 32 + mt * 16 + half * 8 + j;
        if (row < M) C[(size_t)row * Nc + col] = acc[mt][nt2][j] + bv;
      }
    }
  }
}

// =====================================================================
// Small elementwise / graph kernels
// =====================================================================
__global__ void GCL_cvt_bf16(const float* __restrict__ s, uint16_t* __restrict__ d, int n) {
  int i = blockIdx.x * blockDim.x + threadIdx.x;
  if (i < n) d[i] = f2bf(s[i]);
}

// W[K,Nc] (f32) -> Wt[Nc,K] (bf16) transpose+convert
__global__ void GCL_cvtT_bf16(const float* __restrict__ W, uint16_t* __restrict__ Wt,
                              int K, int Nc) {
  int idx = blockIdx.x * blockDim.x + threadIdx.x;
  if (idx < K * Nc) {
    int k = idx / Nc, n = idx % Nc;          // coalesced read over n
    Wt[(size_t)n * K + k] = f2bf(W[idx]);
  }
}

__global__ void GCL_deg_init(float* __restrict__ deg, int n) {
  int i = blockIdx.x * blockDim.x + threadIdx.x;
  if (i < n) deg[i] = 1.0f;                       // self-loop weight
}

__global__ void GCL_deg_accum(const long long* __restrict__ col,
                              const float* __restrict__ w,
                              float* __restrict__ deg, int E) {
  int e = blockIdx.x * blockDim.x + threadIdx.x;
  if (e < E) atomicAdd(&deg[(size_t)col[e]], w[e]);
}

__global__ void GCL_dinv(float* __restrict__ deg, int n) {
  int i = blockIdx.x * blockDim.x + threadIdx.x;
  if (i < n) { float d = deg[i]; deg[i] = d > 0.0f ? rsqrtf(d) : 0.0f; }
}

__global__ void GCL_norm(const long long* __restrict__ ei,
                         const float* __restrict__ w,
                         const float* __restrict__ dinv,
                         float* __restrict__ nrm, int E) {
  int e = blockIdx.x * blockDim.x + threadIdx.x;
  if (e < E) nrm[e] = dinv[(size_t)ei[e]] * w[e] * dinv[(size_t)ei[E + e]];
}

// out[i,f] = dinv[i]^2 * h[i,f] + bias[f]   (self-loop term + layer bias)
__global__ void GCL_agg_init(const float* __restrict__ h,
                             const float* __restrict__ dinv,
                             const float* __restrict__ bias,
                             float* __restrict__ out, int n, int logF) {
  int idx = blockIdx.x * blockDim.x + threadIdx.x;
  if (idx < (n << logF)) {
    int i = idx >> logF;
    int f = idx & ((1 << logF) - 1);
    float di = dinv[i];
    out[idx] = di * di * h[idx] + bias[f];
  }
}

// out[col[e],f] += norm[e] * h[row[e],f]  — L2-resident atomics
__global__ void GCL_agg_scatter(const float* __restrict__ h,
                                const long long* __restrict__ ei,
                                const float* __restrict__ nrm,
                                float* __restrict__ out, int E, int logF) {
  long long idx = (long long)blockIdx.x * blockDim.x + threadIdx.x;
  if (idx >= ((long long)E << logF)) return;
  int e = (int)(idx >> logF);
  int f = (int)(idx & ((1 << logF) - 1));
  size_t r = (size_t)ei[e];
  size_t c = (size_t)ei[E + e];
  atomicAdd(&out[(c << logF) + f], nrm[e] * h[(r << logF) + f]);
}

// relu + f32 -> bf16 (next-layer GEMM input)
__global__ void GCL_relu_bf16(const float* __restrict__ x, uint16_t* __restrict__ d, int n) {
  int i = blockIdx.x * blockDim.x + threadIdx.x;
  if (i < n) d[i] = f2bf(fmaxf(x[i], 0.0f));
}

// =====================================================================
extern "C" void kernel_launch(void* const* d_in, const int* in_sizes, int n_in,
                              void* d_out, int out_size, void* d_ws, size_t ws_size,
                              hipStream_t stream) {
  const float*     x  = (const float*)d_in[0];
  const long long* ei = (const long long*)d_in[1];   // int64 [2,E]
  const float*     ew = (const float*)d_in[2];
  const float*     W1 = (const float*)d_in[3];
  const float*     b1 = (const float*)d_in[4];
  const float*     W2 = (const float*)d_in[5];
  const float*     b2 = (const float*)d_in[6];
  const float*     Wp = (const float*)d_in[7];
  const float*     bp = (const float*)d_in[8];
  float*           out = (float*)d_out;

  const int N = in_sizes[0] / IN_DIM;
  const int E = in_sizes[2];

  // ---- workspace layout (~183 MB incl. aliasing + overread pad) ----
  char* ws = (char*)d_ws;
  size_t off = 0;
  auto wsalloc = [&](size_t bytes) -> void* {
    void* p = (void*)(ws + off);
    off += (bytes + 255) & ~((size_t)255);
    return p;
  };
  float*    deg = (float*)wsalloc((size_t)N * 4);                     // -> dinv in place
  float*    nrm = (float*)wsalloc((size_t)E * 4);
  uint16_t* Wt1 = (uint16_t*)wsalloc((size_t)IN_DIM * HID_DIM * 2);   // [HID,IN]
  uint16_t* Wt2 = (uint16_t*)wsalloc((size_t)HID_DIM * OUT_DIM * 2);  // [OUT,HID]
  uint16_t* Wtp = (uint16_t*)wsalloc((size_t)OUT_DIM * OUT_DIM * 2);  // [OUT,OUT]
  uint16_t* xbf = (uint16_t*)wsalloc((size_t)N * IN_DIM * 2);         // x in bf16
  float*    xw1 = (float*)wsalloc((size_t)N * HID_DIM * 4);           // reused as hw2
  float*    g1  = (float*)wsalloc((size_t)N * HID_DIM * 4);           // reused as g2
  uint16_t* g1b = (uint16_t*)wsalloc((size_t)N * HID_DIM * 2);        // reused as g2b
  (void)wsalloc((size_t)64 * 1024);                                   // A-overread pad
  float*    hw2 = xw1;   // xw1 dead after scatter1
  float*    g2  = g1;    // g1 dead after relu_bf16
  uint16_t* g2b = g1b;   // g1b dead after gemm2

  const int T = 256;
  auto blocks = [](long long n, int t) { return (unsigned)((n + t - 1) / t); };

  // weights -> bf16 transposed [Nc,K]; x -> bf16 (once per call)
  GCL_cvtT_bf16<<<blocks(IN_DIM * HID_DIM, T), T, 0, stream>>>(W1, Wt1, IN_DIM, HID_DIM);
  GCL_cvtT_bf16<<<blocks(HID_DIM * OUT_DIM, T), T, 0, stream>>>(W2, Wt2, HID_DIM, OUT_DIM);
  GCL_cvtT_bf16<<<blocks(OUT_DIM * OUT_DIM, T), T, 0, stream>>>(Wp, Wtp, OUT_DIM, OUT_DIM);
  GCL_cvt_bf16<<<blocks((long long)N * IN_DIM, T), T, 0, stream>>>(x, xbf, N * IN_DIM);

  // gcn_norm
  GCL_deg_init<<<blocks(N, T), T, 0, stream>>>(deg, N);
  GCL_deg_accum<<<blocks(E, T), T, 0, stream>>>(ei + E, ew, deg, E);
  GCL_dinv<<<blocks(N, T), T, 0, stream>>>(deg, N);
  GCL_norm<<<blocks(E, T), T, 0, stream>>>(ei, ew, deg, nrm, E);

  // ---- layer 1: xw1 = x @ W1 ; g1 = D^-1/2 A D^-1/2 xw1 + b1 ; relu ----
  {
    dim3 gr((N + BM - 1) / BM, HID_DIM / BN);
    GCL_gemm_wmma_bf16<<<gr, T, 0, stream>>>(xbf, Wt1, xw1, nullptr, N, IN_DIM, HID_DIM);
  }
  GCL_agg_init<<<blocks((long long)N * HID_DIM, T), T, 0, stream>>>(xw1, deg, b1, g1, N, 8);
  GCL_agg_scatter<<<blocks((long long)E * HID_DIM, T), T, 0, stream>>>(xw1, ei, nrm, g1, E, 8);
  GCL_relu_bf16<<<blocks((long long)N * HID_DIM, T), T, 0, stream>>>(g1, g1b, N * HID_DIM);

  // ---- layer 2: hw2 = relu(g1) @ W2 ; g2 = agg + b2 ----
  {
    dim3 gr((N + BM - 1) / BM, OUT_DIM / BN);
    GCL_gemm_wmma_bf16<<<gr, T, 0, stream>>>(g1b, Wt2, hw2, nullptr, N, HID_DIM, OUT_DIM);
  }
  GCL_agg_init<<<blocks((long long)N * OUT_DIM, T), T, 0, stream>>>(hw2, deg, b2, g2, N, 7);
  GCL_agg_scatter<<<blocks((long long)E * OUT_DIM, T), T, 0, stream>>>(hw2, ei, nrm, g2, E, 7);
  GCL_cvt_bf16<<<blocks((long long)N * OUT_DIM, T), T, 0, stream>>>(g2, g2b, N * OUT_DIM);

  // ---- projection: out = g2 @ Wp + bp ----
  {
    dim3 gr((N + BM - 1) / BM, OUT_DIM / BN);
    GCL_gemm_wmma_bf16<<<gr, T, 0, stream>>>(g2b, Wtp, out, bp, N, OUT_DIM, OUT_DIM);
  }
}
